// LongformerAPTDetector_39917426049455
// MI455X (gfx1250) — matrix operand order repaired and checked by reading, not compiled, not run on hardware
//
#include <hip/hip_runtime.h>
#include <hip/hip_bf16.h>
#include <math.h>

#define BB   64
#define SS   500
#define DIN  64
#define DD   128
#define NH   8
#define HD   16
#define HALF 5
#define WIN  11
#define NL   2
#define DFF  512
#define NTOK (BB * SS)   // 32000
#define LDSR 40          // LDS row stride in halves (32 data + 8 pad, keeps 16B align)

#if __has_builtin(__builtin_amdgcn_global_load_async_to_lds_b128) && \
    __has_builtin(__builtin_amdgcn_s_wait_asynccnt)
#define HAS_ASYNC_LDS 1
#else
#define HAS_ASYNC_LDS 0
#endif

typedef __attribute__((ext_vector_type(16))) __bf16 v16bf;
typedef __attribute__((ext_vector_type(8)))  __bf16 v8bf;
typedef __attribute__((ext_vector_type(8)))  float  v8f;
typedef int v4i32 __attribute__((__vector_size__(16)));

static __device__ __forceinline__ unsigned short f32_bf16(float f) {
  unsigned int u = __float_as_uint(f);
  u += 0x7fffu + ((u >> 16) & 1u);     // round-to-nearest-even
  return (unsigned short)(u >> 16);
}

static __device__ __forceinline__ float wave_sum(float x) {
#pragma unroll
  for (int off = 16; off > 0; off >>= 1) x += __shfl_xor(x, off, 32);
  return x;
}

// ---------------------------------------------------------------------------
// f32 -> bf16 elementwise convert (weights / input only)
// ---------------------------------------------------------------------------
__global__ void cvt_kernel(const float* __restrict__ src,
                           unsigned short* __restrict__ dst, int n) {
  int i = blockIdx.x * blockDim.x + threadIdx.x;
  if (i < n) dst[i] = f32_bf16(src[i]);
}

// ---------------------------------------------------------------------------
// Generic bf16 WMMA GEMM: C[M,N] = A[M,K] @ B[K,N] + bias[N]
//   act=1 -> exact GELU in epilogue; obf=1 -> emit bf16(ushort), else f32.
// Block: 256 threads (8 waves). Block tile 64x64, K-step 32.
// Wave (wm,wn): wm in {0,1} covers 32 M-rows (two 16x16 WMMA tiles),
//               wn in {0..3} covers 16 N-cols.
// A tile staged via GLOBAL_LOAD_ASYNC_TO_LDS_B128 (ASYNCcnt) when available.
// ---------------------------------------------------------------------------
__global__ __launch_bounds__(256) void gemm_bf16_kernel(
    const unsigned short* __restrict__ A, const unsigned short* __restrict__ Bw,
    const float* __restrict__ bias, void* __restrict__ Cout,
    int M, int N, int K, int act, int obf) {
  __shared__ unsigned short Alds[64 * LDSR];
  __shared__ unsigned short Blds[64 * LDSR];   // transposed: [n][k]

  const int tid  = threadIdx.x;
  const int lane = tid & 31;
  const int wave = tid >> 5;
  const int wm = wave >> 2;        // 0..1
  const int wn = wave & 3;         // 0..3
  const int m0 = blockIdx.y * 64;
  const int n0 = blockIdx.x * 64;

  v8f acc0 = {0.f,0.f,0.f,0.f,0.f,0.f,0.f,0.f};
  v8f acc1 = {0.f,0.f,0.f,0.f,0.f,0.f,0.f,0.f};

  // LDS fill assignments: 2048 elems per tile, 8 per thread (16B)
  const int aRow = tid >> 2;            // 0..63
  const int aCol = (tid & 3) * 8;       // 0,8,16,24
  const int bRow = tid >> 3;            // 0..31 (k)
  const int bCol = (tid & 7) * 8;       // 0..56 (n)

  const int lr = lane & 15;
  const int kh = lane >> 4;

  for (int k0 = 0; k0 < K; k0 += 32) {
    // ---- stage A tile (row-major) ----
    {
      const unsigned short* src = &A[(size_t)(m0 + aRow) * K + k0 + aCol];
      unsigned short* dst = &Alds[aRow * LDSR + aCol];
#if HAS_ASYNC_LDS
      __builtin_amdgcn_global_load_async_to_lds_b128(
          (v4i32 __attribute__((address_space(1)))*)src,
          (v4i32 __attribute__((address_space(3)))*)dst, 0, 0);
#else
#pragma unroll
      for (int i = 0; i < 8; ++i) dst[i] = src[i];
#endif
    }
    // ---- stage B tile transposed ([n][k]) ----
    {
      const unsigned short* src = &Bw[(size_t)(k0 + bRow) * N + n0 + bCol];
#pragma unroll
      for (int i = 0; i < 8; ++i) Blds[(bCol + i) * LDSR + bRow] = src[i];
    }
#if HAS_ASYNC_LDS
    __builtin_amdgcn_s_wait_asynccnt(0);
#endif
    __syncthreads();

    union Frag { v16bf v; v8bf h[2]; };
    // B fragment: lane = N col, 16 contiguous K halves (kh selects K 0-15 / 16-31)
    Frag fb;
    {
      const unsigned short* pb = &Blds[(wn * 16 + lr) * LDSR + kh * 16];
      fb.h[0] = *(const v8bf*)pb;
      fb.h[1] = *(const v8bf*)(pb + 8);
    }
    // A fragments: lane = M row; 16 contiguous K halves (kh selects K 0-15 / 16-31)
    Frag fa0, fa1;
    {
      const unsigned short* pa = &Alds[(wm * 32 + lr) * LDSR + kh * 16];
      fa0.h[0] = *(const v8bf*)pa;
      fa0.h[1] = *(const v8bf*)(pa + 8);
      const unsigned short* pa1 = pa + 16 * LDSR;
      fa1.h[0] = *(const v8bf*)pa1;
      fa1.h[1] = *(const v8bf*)(pa1 + 8);
    }
    acc0 = __builtin_amdgcn_wmma_f32_16x16x32_bf16(false, fa0.v, false, fb.v,
                                                   (short)0, acc0, false, false);
    acc1 = __builtin_amdgcn_wmma_f32_16x16x32_bf16(false, fa1.v, false, fb.v,
                                                   (short)0, acc1, false, false);
    __syncthreads();
  }

  // Epilogue. C/D layout: VGPR r, lanes 0-15 -> M=r, lanes 16-31 -> M=8+r; N=lane&15
  const int nCol  = n0 + wn * 16 + lr;
  const int mBase = m0 + wm * 32 + (kh << 3);
  const float bvs = bias[nCol];
  float* Cf = (float*)Cout;
  unsigned short* Cb = (unsigned short*)Cout;
#pragma unroll
  for (int r = 0; r < 8; ++r) {
    float v0 = acc0[r] + bvs;
    float v1 = acc1[r] + bvs;
    if (act) {
      v0 = 0.5f * v0 * (1.0f + erff(v0 * 0.70710678118654752f));
      v1 = 0.5f * v1 * (1.0f + erff(v1 * 0.70710678118654752f));
    }
    size_t i0 = (size_t)(mBase + r) * N + nCol;
    size_t i1 = (size_t)(mBase + 16 + r) * N + nCol;
    if (obf) { Cb[i0] = f32_bf16(v0); Cb[i1] = f32_bf16(v1); }
    else     { Cf[i0] = v0;           Cf[i1] = v1; }
  }
}

// ---------------------------------------------------------------------------
// Embedding epilogue: h = LN(gemm_out + pos_emb[s+2] + tt_emb); emit f32+bf16
// One wave per token row (D=128 -> 4 elems/lane).
// ---------------------------------------------------------------------------
__global__ __launch_bounds__(256) void embed_ln_kernel(
    const float* __restrict__ a, const float* __restrict__ pos,
    const float* __restrict__ tt, const float* __restrict__ g,
    const float* __restrict__ b, float* __restrict__ hf,
    unsigned short* __restrict__ hb) {
  int row  = blockIdx.x * 8 + (threadIdx.x >> 5);
  int lane = threadIdx.x & 31;
  int s = row % SS;
  const float* ar = a + (size_t)row * DD;
  const float* pr = pos + (size_t)(s + 2) * DD;
  float x[4];
#pragma unroll
  for (int i = 0; i < 4; ++i) { int d = lane + 32 * i; x[i] = ar[d] + pr[d] + tt[d]; }
  float mean = wave_sum(x[0] + x[1] + x[2] + x[3]) * (1.f / DD);
  float vs = 0.f;
#pragma unroll
  for (int i = 0; i < 4; ++i) { float t = x[i] - mean; vs += t * t; }
  float inv = 1.f / sqrtf(wave_sum(vs) * (1.f / DD) + 1e-12f);
#pragma unroll
  for (int i = 0; i < 4; ++i) {
    int d = lane + 32 * i;
    float y = (x[i] - mean) * inv * g[d] + b[d];
    hf[(size_t)row * DD + d] = y;
    hb[(size_t)row * DD + d] = f32_bf16(y);
  }
}

// ---------------------------------------------------------------------------
// Residual + LN (in-place on residual stream): h = LN(a + h); emit f32+bf16
// ---------------------------------------------------------------------------
__global__ __launch_bounds__(256) void add_ln_kernel(
    const float* __restrict__ a, float* __restrict__ hf,
    const float* __restrict__ g, const float* __restrict__ b,
    unsigned short* __restrict__ hb) {
  int row  = blockIdx.x * 8 + (threadIdx.x >> 5);
  int lane = threadIdx.x & 31;
  const float* ar = a + (size_t)row * DD;
  float* hr = hf + (size_t)row * DD;
  float x[4];
#pragma unroll
  for (int i = 0; i < 4; ++i) { int d = lane + 32 * i; x[i] = ar[d] + hr[d]; }
  float mean = wave_sum(x[0] + x[1] + x[2] + x[3]) * (1.f / DD);
  float vs = 0.f;
#pragma unroll
  for (int i = 0; i < 4; ++i) { float t = x[i] - mean; vs += t * t; }
  float inv = 1.f / sqrtf(wave_sum(vs) * (1.f / DD) + 1e-12f);
#pragma unroll
  for (int i = 0; i < 4; ++i) {
    int d = lane + 32 * i;
    float y = (x[i] - mean) * inv * g[d] + b[d];
    hr[d] = y;
    hb[(size_t)row * DD + d] = f32_bf16(y);
  }
}

// ---------------------------------------------------------------------------
// Windowed attention: one thread per (b, s, head). WIN=11, HD=16 in f32.
// Emits bf16 context directly (feeds the Wo GEMM).
// ---------------------------------------------------------------------------
__global__ __launch_bounds__(256) void attn_kernel(
    const float* __restrict__ q, const float* __restrict__ k,
    const float* __restrict__ v, unsigned short* __restrict__ ctxb) {
  int idx = blockIdx.x * blockDim.x + threadIdx.x;   // [0, BB*SS*NH)
  int hh = idx % NH;
  int s  = (idx / NH) % SS;
  int b  = idx / (NH * SS);
  size_t rowq = ((size_t)b * SS + s) * DD + hh * HD;
  float qreg[HD];
#pragma unroll
  for (int d = 0; d < HD; ++d) qreg[d] = q[rowq + d] * 0.25f;  // 1/sqrt(16)
  float sc[WIN];
  float mx = -1e30f;
#pragma unroll
  for (int w = 0; w < WIN; ++w) {
    int t = s + w - HALF;
    float dot = -1e30f;
    if (t >= 0 && t < SS) {
      size_t rk = ((size_t)b * SS + t) * DD + hh * HD;
      dot = 0.f;
#pragma unroll
      for (int d = 0; d < HD; ++d) dot += qreg[d] * k[rk + d];
    }
    sc[w] = dot;
    mx = fmaxf(mx, dot);
  }
  float den = 0.f;
#pragma unroll
  for (int w = 0; w < WIN; ++w) { sc[w] = __expf(sc[w] - mx); den += sc[w]; }
  float inv = 1.f / den;
  float acc[HD];
#pragma unroll
  for (int d = 0; d < HD; ++d) acc[d] = 0.f;
#pragma unroll
  for (int w = 0; w < WIN; ++w) {
    int t = s + w - HALF;
    if (t >= 0 && t < SS) {
      size_t rv = ((size_t)b * SS + t) * DD + hh * HD;
#pragma unroll
      for (int d = 0; d < HD; ++d) acc[d] += sc[w] * v[rv + d];
    }
  }
#pragma unroll
  for (int d = 0; d < HD; ++d) ctxb[rowq + d] = f32_bf16(acc[d] * inv);
}

// ---------------------------------------------------------------------------
// Mean-pool over S then 128->2 classifier. One block per batch element.
// ---------------------------------------------------------------------------
__global__ __launch_bounds__(128) void pool_cls_kernel(
    const float* __restrict__ h, const float* __restrict__ Wc,
    const float* __restrict__ bc, float* __restrict__ out) {
  __shared__ float pooled[DD];
  int b = blockIdx.x;
  int d = threadIdx.x;
  float s = 0.f;
  for (int t = 0; t < SS; ++t) s += h[((size_t)b * SS + t) * DD + d];
  pooled[d] = s * (1.0f / SS);
  __syncthreads();
  if (d < 2) {
    float a = bc[d];
    for (int j = 0; j < DD; ++j) a += pooled[j] * Wc[j * 2 + d];
    out[b * 2 + d] = a;
  }
}

// ---------------------------------------------------------------------------
extern "C" void kernel_launch(void* const* d_in, const int* in_sizes, int n_in,
                              void* d_out, int out_size, void* d_ws, size_t ws_size,
                              hipStream_t stream) {
  (void)in_sizes; (void)n_in; (void)out_size; (void)ws_size;
  const float* x    = (const float*)d_in[0];
  const float* W_in = (const float*)d_in[1];
  const float* b_in = (const float*)d_in[2];
  const float* pos  = (const float*)d_in[3];
  const float* tt   = (const float*)d_in[4];
  const float* lng  = (const float*)d_in[5];
  const float* lnb  = (const float*)d_in[6];
  const float* Wq   = (const float*)d_in[7];
  const float* bq   = (const float*)d_in[8];
  const float* Wk   = (const float*)d_in[9];
  const float* bk   = (const float*)d_in[10];
  const float* Wv   = (const float*)d_in[11];
  const float* bv   = (const float*)d_in[12];
  const float* Wo   = (const float*)d_in[13];
  const float* bo   = (const float*)d_in[14];
  const float* ln1g = (const float*)d_in[15];
  const float* ln1b = (const float*)d_in[16];
  const float* Wi   = (const float*)d_in[17];
  const float* bi   = (const float*)d_in[18];
  const float* Wo2  = (const float*)d_in[19];
  const float* bo2  = (const float*)d_in[20];
  const float* ln2g = (const float*)d_in[21];
  const float* ln2b = (const float*)d_in[22];
  const float* Wcls = (const float*)d_in[23];
  const float* bcls = (const float*)d_in[24];
  float* out = (float*)d_out;

  char* p = (char*)d_ws;
  auto carve = [&](size_t bytes) -> char* {
    char* r = p; p += (bytes + 255) & ~(size_t)255; return r;
  };
  unsigned short* Winb = (unsigned short*)carve((size_t)DIN * DD * 2);
  unsigned short* Wqb  = (unsigned short*)carve((size_t)NL * DD * DD * 2);
  unsigned short* Wkb  = (unsigned short*)carve((size_t)NL * DD * DD * 2);
  unsigned short* Wvb  = (unsigned short*)carve((size_t)NL * DD * DD * 2);
  unsigned short* Wob  = (unsigned short*)carve((size_t)NL * DD * DD * 2);
  unsigned short* Wib  = (unsigned short*)carve((size_t)NL * DD * DFF * 2);
  unsigned short* Wo2b = (unsigned short*)carve((size_t)NL * DFF * DD * 2);
  unsigned short* xb   = (unsigned short*)carve((size_t)NTOK * DIN * 2);
  float*          hf   = (float*)carve((size_t)NTOK * DD * 4);
  unsigned short* hb   = (unsigned short*)carve((size_t)NTOK * DD * 2);
  float*          qf   = (float*)carve((size_t)NTOK * DD * 4);
  float*          kf   = (float*)carve((size_t)NTOK * DD * 4);
  float*          vf   = (float*)carve((size_t)NTOK * DD * 4);
  unsigned short* cb   = (unsigned short*)carve((size_t)NTOK * DD * 2);
  float*          dtmp = (float*)carve((size_t)NTOK * DD * 4);
  unsigned short* ffb  = (unsigned short*)carve((size_t)NTOK * DFF * 2);

  auto cvt = [&](const float* s, unsigned short* d, int n) {
    cvt_kernel<<<(n + 255) / 256, 256, 0, stream>>>(s, d, n);
  };
  cvt(W_in, Winb, DIN * DD);
  cvt(Wq,  Wqb,  NL * DD * DD);
  cvt(Wk,  Wkb,  NL * DD * DD);
  cvt(Wv,  Wvb,  NL * DD * DD);
  cvt(Wo,  Wob,  NL * DD * DD);
  cvt(Wi,  Wib,  NL * DD * DFF);
  cvt(Wo2, Wo2b, NL * DFF * DD);
  cvt(x,   xb,   NTOK * DIN);

  // Embedding: dtmp = x @ W_in + b_in ; h = LN(dtmp + pos + tt)
  gemm_bf16_kernel<<<dim3(DD / 64, NTOK / 64), 256, 0, stream>>>(
      xb, Winb, b_in, dtmp, NTOK, DD, DIN, 0, 0);
  embed_ln_kernel<<<NTOK / 8, 256, 0, stream>>>(dtmp, pos, tt, lng, lnb, hf, hb);

  for (int l = 0; l < NL; ++l) {
    gemm_bf16_kernel<<<dim3(DD / 64, NTOK / 64), 256, 0, stream>>>(
        hb, Wqb + (size_t)l * DD * DD, bq + l * DD, qf, NTOK, DD, DD, 0, 0);
    gemm_bf16_kernel<<<dim3(DD / 64, NTOK / 64), 256, 0, stream>>>(
        hb, Wkb + (size_t)l * DD * DD, bk + l * DD, kf, NTOK, DD, DD, 0, 0);
    gemm_bf16_kernel<<<dim3(DD / 64, NTOK / 64), 256, 0, stream>>>(
        hb, Wvb + (size_t)l * DD * DD, bv + l * DD, vf, NTOK, DD, DD, 0, 0);

    // windowed attention -> bf16 context
    attn_kernel<<<(BB * SS * NH) / 256, 256, 0, stream>>>(qf, kf, vf, cb);

    gemm_bf16_kernel<<<dim3(DD / 64, NTOK / 64), 256, 0, stream>>>(
        cb, Wob + (size_t)l * DD * DD, bo + l * DD, dtmp, NTOK, DD, DD, 0, 0);
    add_ln_kernel<<<NTOK / 8, 256, 0, stream>>>(dtmp, hf, ln1g + l * DD,
                                                ln1b + l * DD, hb);

    // FFN: GELU fused + bf16 output straight into ffb (no cvt pass)
    gemm_bf16_kernel<<<dim3(DFF / 64, NTOK / 64), 256, 0, stream>>>(
        hb, Wib + (size_t)l * DD * DFF, bi + l * DFF, ffb, NTOK, DFF, DD, 1, 1);
    gemm_bf16_kernel<<<dim3(DD / 64, NTOK / 64), 256, 0, stream>>>(
        ffb, Wo2b + (size_t)l * DFF * DD, bo2 + l * DD, dtmp, NTOK, DD, DFF, 0, 0);
    add_ln_kernel<<<NTOK / 8, 256, 0, stream>>>(dtmp, hf, ln2g + l * DD,
                                                ln2b + l * DD, hb);
  }

  pool_cls_kernel<<<BB, 128, 0, stream>>>(hf, Wcls, bcls, out);
}
